// QuantLinear_9534827397281
// MI455X (gfx1250) — matrix operand order repaired and checked by reading, not compiled, not run on hardware
//
#include <hip/hip_runtime.h>

typedef __attribute__((ext_vector_type(16))) _Float16 v16h;
typedef __attribute__((ext_vector_type(8)))  _Float16 v8h;
typedef __attribute__((ext_vector_type(8)))  float    v8f;

#define M_DIM  8192
#define K_DIM  4096
#define N_DIM  4096
// GROUPS = 32, group_size = 128  ->  g = k >> 7

#define BM  128
#define BN  128
#define BK  32
#define BKP 40          // padded k-stride in halves (+16B) to stagger LDS banks
#define KTILES (K_DIM / BK)

// CDNA5 async global->LDS copy (ASYNCcnt-tracked, no VGPR round trip).
// LDS offset = low 32 bits of the generic pointer (aperture tag lives in the
// high dword; flat->LDS mapping truncates to addr[31:0]).
__device__ __forceinline__ void async_load_b128(const void* gsrc, void* lds_dst) {
  unsigned loff = (unsigned)(unsigned long long)lds_dst;
  asm volatile("global_load_async_to_lds_b128 %0, %1, off"
               :: "v"(loff), "v"(gsrc)
               : "memory");
}

__device__ __forceinline__ void wait_asynccnt0() {
  asm volatile("s_wait_asynccnt 0" ::: "memory");
}

__global__ __launch_bounds__(256)
void exl2_qgemm_wmma(const _Float16* __restrict__ x,
                     const int*      __restrict__ q_weight,   // [K/8, N] int32 (8 nibbles along K)
                     const int*      __restrict__ q_scale,    // [G, N/8] int32 (8 nibbles along N)
                     const _Float16* __restrict__ q_scale_max,// [G]
                     const int*      __restrict__ q_invperm,  // [K]
                     const _Float16* __restrict__ bias,       // [N]
                     _Float16*       __restrict__ out)        // [M, N]
{
  __shared__ __align__(16) _Float16 sA[BM * BKP];  // x tile, row-major [m][k]
  __shared__ __align__(16) _Float16 sB[BN * BKP];  // dequantized W^T tile [n][k]

  const int tid  = threadIdx.x;
  const int lane = tid & 31;
  const int lm   = lane & 15;          // row/col within a 16x16 WMMA tile
  const int wave = tid >> 5;           // 0..7
  const int wm   = wave >> 1;          // 0..3 -> 32-row slab of block tile
  const int wn   = wave & 1;           // 0..1 -> 64-col slab of block tile

  const int m_base = blockIdx.y * BM;
  const int n_base = blockIdx.x * BN;

  v8f acc[2][4];
#pragma unroll
  for (int mt = 0; mt < 2; ++mt)
#pragma unroll
    for (int nt = 0; nt < 4; ++nt)
      acc[mt][nt] = {};

  // staging work split (fixed per thread across K-tiles)
  const int jj = tid >> 3;             // 0..31 : k within tile handled by this thread
  const int n0 = (tid & 7) * 16;       // 16 columns per thread

  for (int kt = 0; kt < KTILES; ++kt) {
    const int k0 = kt * BK;

    // ---- stage x tile asynchronously: 128 rows x 32 halves = 512 x 16B
    //      chunks, 2 per thread, straight into LDS (no VGPR round trip) ----
#pragma unroll
    for (int c = 0; c < 2; ++c) {
      const int chunk = tid + c * 256;
      const int row   = chunk >> 2;
      const int off   = (chunk & 3) * 8;
      async_load_b128(x + (size_t)(m_base + row) * K_DIM + k0 + off,
                      &sA[row * BKP + off]);
    }

    // ---- dequantize permuted weight rows into sB[n][k] (overlaps with the
    //      async x-tile copy) ----
    {
      const int r  = q_invperm[k0 + jj];     // physical weight row for logical k
      const int g  = r >> 7;                 // group (group_size = 128)
      const int sh = (r & 7) * 4;            // nibble position along K
      const float smax = (float)q_scale_max[g] * (1.0f / 256.0f);
      const int* wrow = q_weight + (size_t)(r >> 3) * N_DIM + (n_base + n0);
      const int* srow = q_scale + g * (N_DIM / 8) + ((n_base + n0) >> 3);
      if (kt + 1 < KTILES)
        __builtin_prefetch(q_invperm + k0 + BK + jj, 0, 0);  // global_prefetch_b8
#pragma unroll
      for (int h = 0; h < 2; ++h) {
        const int  sdw = srow[h];
        const int4 wa  = *(const int4*)(wrow + h * 8);
        const int4 wb  = *(const int4*)(wrow + h * 8 + 4);
        const int  w[8] = {wa.x, wa.y, wa.z, wa.w, wb.x, wb.y, wb.z, wb.w};
#pragma unroll
        for (int c = 0; c < 8; ++c) {
          const int   w4    = (w[c] >> sh) & 0xF;
          const int   s4    = (sdw >> (c * 4)) & 0xF;
          const float scale = (float)((s4 + 1) * (s4 + 1)) * smax;
          sB[(n0 + h * 8 + c) * BKP + jj] =
              (_Float16)((float)(w4 - 8) * scale);
        }
      }
    }

    // drain this wave's async copies, then workgroup barrier
    wait_asynccnt0();
    __syncthreads();

    // ---- B fragments: lane = column (wn*64 + nt*16 + lm), contiguous K.
    //      16-bit B 32x16: lanes 0-15 hold K=0..15, lanes 16-31 hold K=16..31.
    const int kbB = (lane >> 4) * 16;
    v16h bfrag[4];
#pragma unroll
    for (int nt = 0; nt < 4; ++nt) {
      const _Float16* bc = &sB[(wn * 64 + nt * 16 + lm) * BKP + kbB];
      const v8h lo = *(const v8h*)(bc);
      const v8h hi = *(const v8h*)(bc + 8);
      bfrag[nt] = __builtin_shufflevector(lo, hi,
          0,1,2,3,4,5,6,7,8,9,10,11,12,13,14,15);
    }

    // ---- A fragments + WMMA. 16-bit A 16x32: lane row = lm,
    //      contiguous K: lanes 0-15 hold K=0..15, lanes 16-31 hold K=16..31.
    const int kbA = (lane >> 4) * 16;
#pragma unroll
    for (int mt = 0; mt < 2; ++mt) {
      const _Float16* ar = &sA[(wm * 32 + mt * 16 + lm) * BKP + kbA];
      const v8h lo = *(const v8h*)(ar);
      const v8h hi = *(const v8h*)(ar + 8);
      const v16h afrag = __builtin_shufflevector(lo, hi,
          0,1,2,3,4,5,6,7,8,9,10,11,12,13,14,15);
#pragma unroll
      for (int nt = 0; nt < 4; ++nt) {
        acc[mt][nt] = __builtin_amdgcn_wmma_f32_16x16x32_f16(
            /*neg_a=*/false, afrag, /*neg_b=*/false, bfrag[nt],
            /*c_mod=*/(short)0, acc[mt][nt],
            /*reuse_a=*/false, /*reuse_b=*/false);
      }
    }

    __syncthreads();
  }

  // ---- epilogue: bias add (f32) + fp16 store.
  //      C/D layout: VGPR v -> M = v + (lane<16 ? 0 : 8), N = lane%16.
  const int mo = (lane >> 4) * 8;
#pragma unroll
  for (int mt = 0; mt < 2; ++mt) {
#pragma unroll
    for (int nt = 0; nt < 4; ++nt) {
      const int gm = m_base + wm * 32 + mt * 16 + mo;
      const int gn = n_base + wn * 64 + nt * 16 + lm;
      const float bf = (float)bias[gn];
      const v8f c = acc[mt][nt];
#pragma unroll
      for (int v = 0; v < 8; ++v)
        out[(size_t)(gm + v) * N_DIM + gn] = (_Float16)(c[v] + bf);
    }
  }
}

extern "C" void kernel_launch(void* const* d_in, const int* in_sizes, int n_in,
                              void* d_out, int out_size, void* d_ws, size_t ws_size,
                              hipStream_t stream) {
  const _Float16* x   = (const _Float16*)d_in[0];
  const int*      qw  = (const int*)d_in[1];
  const int*      qs  = (const int*)d_in[2];
  const _Float16* qsm = (const _Float16*)d_in[3];
  const int*      inv = (const int*)d_in[4];
  const _Float16* bs  = (const _Float16*)d_in[5];
  _Float16*       out = (_Float16*)d_out;

  dim3 grid(N_DIM / BN, M_DIM / BM);   // (32, 64) = 2048 blocks
  exl2_qgemm_wmma<<<grid, 256, 0, stream>>>(x, qw, qs, qsm, inv, bs, out);

  (void)in_sizes; (void)n_in; (void)out_size; (void)d_ws; (void)ws_size;
}
